// LocalDecoder_3410204033211
// MI455X (gfx1250) — matrix operand order repaired and checked by reading, not compiled, not run on hardware
//
#include <hip/hip_runtime.h>

typedef __attribute__((ext_vector_type(16))) _Float16     v16h;
typedef __attribute__((ext_vector_type(8)))  float        v8f;
typedef __attribute__((ext_vector_type(4)))  unsigned int u32x4;
typedef __attribute__((ext_vector_type(4)))  float        f32x4;

// Problem constants
constexpr int NPTS  = 4 * 131072;   // B*N
constexpr int NFRAG = 52;           // A-matrix weight fragments (16x32 f16 tiles)
constexpr int BIASU = NFRAG * 256;  // u32 offset of f32 bias region in d_ws
// bias region (floats): [0..159]=bc(+bp for blk0), [160..319]=b0, [320..479]=b1,
//                       [480..511]=Wo, [512]=bo
constexpr int WAVES = 8;            // waves per workgroup
constexpr int TPW   = 8;            // 16-point tiles per wave
constexpr int CST   = 68;           // u32 stride per point, feature LDS (128 halves + pad)
constexpr int AST   = 20;           // u32 stride per point, activation LDS (32 halves + pad)

union HFrag { v16h v; u32x4 q[2]; };

__device__ inline float relu(float x) { return x > 0.f ? x : 0.f; }  // -> v_max_num_f32

__device__ inline unsigned int pk2(float a, float b) {
  union { _Float16 h[2]; unsigned int u; } x;
  x.h[0] = (_Float16)a; x.h[1] = (_Float16)b;
  return x.u;
}

__device__ inline u32x4 reluPack(v8f t) {
  u32x4 r;
  r[0] = pk2(relu(t[0]), relu(t[1]));
  r[1] = pk2(relu(t[2]), relu(t[3]));
  r[2] = pk2(relu(t[4]), relu(t[5]));
  r[3] = pk2(relu(t[6]), relu(t[7]));
  return r;
}

__device__ inline v8f wmma_f16(v16h a, v16h b, v8f c) {
  return __builtin_amdgcn_wmma_f32_16x16x32_f16(false, a, false, b, (short)0, c,
                                                false, false);
}

// ---------------------------------------------------------------------------
// Prep: swizzle weights into WMMA A-fragment layout (f16) + bias arrays (f32).
// Frag order per block: [Wc chunks][fo=0,1] ..., then W0 fo=0,1, then W1 fo=0,1.
// Block 0 has a 4th "chunk" = [Wp rows 0..2; zeros] so p@Wp folds into the GEMM.
// ---------------------------------------------------------------------------
__global__ void __launch_bounds__(256)
prep_kernel(const float* __restrict__ Wp, const float* __restrict__ Wc,
            const float* __restrict__ bp, const float* __restrict__ bc,
            const float* __restrict__ W0, const float* __restrict__ b0,
            const float* __restrict__ W1, const float* __restrict__ b1,
            const float* __restrict__ Wo, const float* __restrict__ bo,
            unsigned int* __restrict__ ws)
{
  int t = blockIdx.x * 256 + threadIdx.x;
  if (t < NFRAG * 256) {
    int frag = t >> 8;
    int lane = (t >> 3) & 31;
    int j    = t & 7;
    int blk, rem;
    if (frag < 12) { blk = 0; rem = frag; }
    else           { blk = 1 + (frag - 12) / 10; rem = (frag - 12) % 10; }
    const int nc = (blk == 0) ? 4 : 3;
    int kind, ch = 0, fo;
    if (rem < nc * 2)          { kind = 0; ch = rem >> 1; fo = rem & 1; }
    else if (rem < nc * 2 + 2) { kind = 1; fo = rem - nc * 2; }
    else                       { kind = 2; fo = rem - nc * 2 - 2; }
    const int m = lane & 15, hi = lane >> 4;
    float vals[2];
    #pragma unroll
    for (int tt = 0; tt < 2; ++tt) {
      int e = 2 * j + tt;
      int K = e + 16 * hi;                        // contiguous k order (matches B LDS reads)
      int fout = fo * 16 + m;
      float v = 0.f;
      if (kind == 0) {
        if (ch < 3)      v = Wc[(blk * 96 + ch * 32 + K) * 32 + fout];
        else if (K < 3)  v = Wp[K * 32 + fout];   // p-projection chunk (blk0 only)
      } else if (kind == 1) {
        v = W0[(blk * 32 + K) * 32 + fout];
      } else {
        v = W1[(blk * 32 + K) * 32 + fout];
      }
      vals[tt] = v;
    }
    ws[frag * 256 + lane * 8 + j] = pk2(vals[0], vals[1]);
  } else {
    int r = t - NFRAG * 256;
    float* wb = (float*)(ws + BIASU);
    if (r < 160)      { int blk = r >> 5, f = r & 31;
                        wb[r] = bc[blk * 32 + f] + (blk == 0 ? bp[f] : 0.f); }
    else if (r < 320) { wb[r] = b0[r - 160]; }
    else if (r < 480) { wb[r] = b1[r - 320]; }
    else if (r < 512) { wb[r] = Wo[r - 480]; }
    else if (r == 512){ wb[r] = bo[0]; }
  }
}

// ---------------------------------------------------------------------------
// Main kernel: per-wave 16-point tiles; bilinear gather -> LDS B-fragments ->
// 52 WMMAs per tile; activations relu+f16-packed through per-wave LDS.
// ---------------------------------------------------------------------------
__global__ void __launch_bounds__(256)
decoder_kernel(const float* __restrict__ xw,
               const float* __restrict__ cxy,
               const float* __restrict__ cxz,
               const float* __restrict__ cyz,
               const unsigned int* __restrict__ wsu,
               float* __restrict__ out)
{
  __shared__ unsigned int ldsC[WAVES * 16 * CST];   // features (96) + [p,0..] pad
  __shared__ unsigned int ldsA[WAVES * 16 * AST];   // relu activations

  const int wave = threadIdx.x >> 5;
  const int lane = threadIdx.x & 31;
  const int pt   = lane & 15;
  const int hi   = lane >> 4;

  unsigned int* myC = ldsC + wave * (16 * CST);
  unsigned int* myA = ldsA + wave * (16 * AST);

  for (int t = 0; t < TPW; ++t) {
    // Anti-LICM opacity on an *integer* offset so wsu keeps its global
    // address-space inference (global_load_b128, not flat_load_b128).
    int wsOff = 0;
    asm volatile("" : "+s"(wsOff));
    const unsigned int* wf = wsu + wsOff;
    const float* wb = (const float*)(wf + BIASU);

    auto loadA = [&](int frag) -> v16h {
      HFrag f;
      const u32x4* p = (const u32x4*)(wf + frag * 256 + lane * 8);
      f.q[0] = p[0]; f.q[1] = p[1];
      return f.v;
    };
    auto loadBc = [&](int ch) -> v16h {              // B-frag from feature LDS
      HFrag f;
      f.q[0] = *(const u32x4*)&myC[pt * CST + ch * 16 + hi * 8];
      f.q[1] = *(const u32x4*)&myC[pt * CST + ch * 16 + hi * 8 + 4];
      return f.v;
    };
    auto loadBa = [&]() -> v16h {                    // B-frag from activation LDS
      HFrag f;
      f.q[0] = *(const u32x4*)&myA[pt * AST + hi * 8];
      f.q[1] = *(const u32x4*)&myA[pt * AST + hi * 8 + 4];
      return f.v;
    };
    auto storeAct = [&](v8f t0, v8f t1) {            // D-layout -> relu f16 pairs
      *(u32x4*)&myA[pt * AST + 4 * hi]     = reluPack(t0);
      *(u32x4*)&myA[pt * AST + 8 + 4 * hi] = reluPack(t1);
    };
    auto loadBias = [&](int foff) -> v8f {           // C operand = bias broadcast
      const f32x4* p = (const f32x4*)(wb + foff + hi * 8);
      f32x4 a = p[0], b = p[1];
      v8f c;
      c[0]=a[0]; c[1]=a[1]; c[2]=a[2]; c[3]=a[3];
      c[4]=b[0]; c[5]=b[1]; c[6]=b[2]; c[7]=b[3];
      return c;
    };

    const int tile  = (blockIdx.x * WAVES + wave) * TPW + t;
    const int gp    = tile * 16 + pt;
    const int batch = gp >> 17;                      // N = 2^17
    const float px = xw[gp * 3 + 0];
    const float py = xw[gp * 3 + 1];
    const float pz = xw[gp * 3 + 2];

    // pad slots 48..63 of feature row: [px,py][pz,0][zeros...]
    u32x4 z4 = {0u, 0u, 0u, 0u};
    if (hi == 0) {
      myC[pt * CST + 48] = pk2(px, py);
      myC[pt * CST + 49] = pk2(pz, 0.f);
      myC[pt * CST + 50] = 0u;
      myC[pt * CST + 51] = 0u;
      *(u32x4*)&myC[pt * CST + 52] = z4;
    } else {
      *(u32x4*)&myC[pt * CST + 56] = z4;
      *(u32x4*)&myC[pt * CST + 60] = z4;
    }

    // ---- bilinear tri-plane sampling (channel pairs -> packed f16 in LDS) ----
    auto samp = [&](const float* __restrict__ pl, float u, float v,
                    int q, int sl0, int nsl) {
      const float INV = 1.0f / 1.001f;
      float xu = fminf(fmaxf(u * INV + 0.5f, 0.f), 0.999999f);
      float yv = fminf(fmaxf(v * INV + 0.5f, 0.f), 0.999999f);
      float x = xu * 127.f, y = yv * 127.f;
      float xf = floorf(x), yf = floorf(y);
      float wx = x - xf, wy = y - yf;
      int ix = (int)xf, iy = (int)yf;                // ix,iy <= 126 => +1 safe
      const float* base = pl + ((batch << 19) + iy * 128 + ix);
      #pragma unroll
      for (int i = 0; i < nsl; ++i) {
        int sl = sl0 + i;
        const float* cp = base + sl * 32768;         // 2 channels per slot
        float a00 = cp[0],     a01 = cp[1],     a10 = cp[128],   a11 = cp[129];
        float b00 = cp[16384], b01 = cp[16385], b10 = cp[16512], b11 = cp[16513];
        float ta = a00 + wx * (a01 - a00), ba = a10 + wx * (a11 - a10);
        float ra = ta + wy * (ba - ta);
        float tb = b00 + wx * (b01 - b00), bb = b10 + wx * (b11 - b10);
        float rb = tb + wy * (bb - tb);
        myC[pt * CST + q * 16 + sl] = pk2(ra, rb);
      }
    };
    if (hi == 0) samp(cxy, px, py, 0, 0, 16);        // plane xy: 16 slots
    samp(cxz, px, pz, 1, hi * 8, 8);                 // plane xz: split 8+8
    if (hi == 1) samp(cyz, py, pz, 2, 0, 16);        // plane yz: 16 slots

    // ---- residual MLP via WMMA (net^T: f_out on M, pt on N) ----
    v8f n0 = {0,0,0,0,0,0,0,0};
    v8f n1 = {0,0,0,0,0,0,0,0};
    #pragma unroll
    for (int blk = 0; blk < 5; ++blk) {
      const int fb = (blk == 0) ? 0 : (12 + 10 * (blk - 1));
      const int nc = (blk == 0) ? 4 : 3;             // blk0: +p-projection chunk
      v8f c0, c1;
      if (blk == 0) {                                // net==0: bias is the init
        c0 = loadBias(0);
        c1 = loadBias(16);
      } else {
        c0 = n0 + loadBias(blk * 32);
        c1 = n1 + loadBias(blk * 32 + 16);
      }
      #pragma unroll
      for (int ch = 0; ch < nc; ++ch) {
        v16h bf = loadBc(ch);
        c0 = wmma_f16(loadA(fb + ch * 2 + 0), bf, c0);
        c1 = wmma_f16(loadA(fb + ch * 2 + 1), bf, c1);
      }
      n0 = c0; n1 = c1;                              // net += c@Wc + bias
      storeAct(n0, n1);                              // relu(net) -> LDS (f16)
      v16h hb = loadBa();
      v8f h0 = wmma_f16(loadA(fb + nc * 2 + 0), hb, loadBias(160 + blk * 32));
      v8f h1 = wmma_f16(loadA(fb + nc * 2 + 1), hb, loadBias(160 + blk * 32 + 16));
      storeAct(h0, h1);                              // relu(h) -> LDS
      v16h db = loadBa();
      n0 = wmma_f16(loadA(fb + nc * 2 + 2), db, n0 + loadBias(320 + blk * 32));
      n1 = wmma_f16(loadA(fb + nc * 2 + 3), db, n1 + loadBias(320 + blk * 32 + 16));
    }

    // ---- output projection: relu(net) @ Wo + bo, xor-16 lane reduction ----
    const float* wo = wb + 480;
    float s = 0.f;
    #pragma unroll
    for (int r = 0; r < 8; ++r) {
      s += relu(n0[r]) * wo[8 * hi + r];
      s += relu(n1[r]) * wo[16 + 8 * hi + r];
    }
    union { float f; int i; } cvi; cvi.f = s;
    int sw = __builtin_amdgcn_ds_swizzle(cvi.i, 0x401F);  // SWAPX16
    union { int i; float f; } cvo; cvo.i = sw;
    s += cvo.f;
    if (hi == 0) out[gp] = s + wb[512];
  }
}

extern "C" void kernel_launch(void* const* d_in, const int* in_sizes, int n_in,
                              void* d_out, int out_size, void* d_ws, size_t ws_size,
                              hipStream_t stream) {
  const float* xw  = (const float*)d_in[0];
  const float* cxy = (const float*)d_in[1];
  const float* cxz = (const float*)d_in[2];
  const float* cyz = (const float*)d_in[3];
  const float* Wp  = (const float*)d_in[4];
  const float* bp  = (const float*)d_in[5];
  const float* Wc  = (const float*)d_in[6];
  const float* bc  = (const float*)d_in[7];
  const float* W0  = (const float*)d_in[8];
  const float* b0  = (const float*)d_in[9];
  const float* W1  = (const float*)d_in[10];
  const float* b1  = (const float*)d_in[11];
  const float* Wo  = (const float*)d_in[12];
  const float* bo  = (const float*)d_in[13];
  unsigned int* ws = (unsigned int*)d_ws;

  // (NFRAG*256 + 513) elements to produce
  prep_kernel<<<55, 256, 0, stream>>>(Wp, Wc, bp, bc, W0, b0, W1, b1, Wo, bo, ws);

  // 512 WGs * 8 waves * 8 tiles * 16 pts = 524288 points, exact cover
  decoder_kernel<<<512, 256, 0, stream>>>(xw, cxy, cxz, cyz, ws, (float*)d_out);
}
